// GraphEmbeddingModel_48241072668954
// MI455X (gfx1250) — matrix operand order repaired and checked by reading, not compiled, not run on hardware
//
#include <hip/hip_runtime.h>
#include <hip/hip_bf16.h>

// ---------------------------------------------------------------------------
// GAT 4-layer pipeline for gfx1250 (MI455X).
// GEMMs via v_wmma_f32_16x16x32_bf16 (templated tiles -> static VGPR accums);
// edge softmax-aggregation via f32 atomics.
// ---------------------------------------------------------------------------

#define NNODES 50000
#define NEDGES 800000
#define NETOT  (NEDGES + NNODES)   // with self loops
#define NGRAPH 64

typedef __attribute__((ext_vector_type(16))) __bf16 v16bf;
typedef __attribute__((ext_vector_type(8)))  float  v8f;

union Frag16 { v16bf v; uint4 q[2]; };

// ---- order-preserving float<->uint encoding for atomic max over signed f32 --
__device__ __forceinline__ unsigned enc_f32(float f) {
    unsigned u = __float_as_uint(f);
    return (u & 0x80000000u) ? ~u : (u | 0x80000000u);
}
__device__ __forceinline__ float dec_f32(unsigned u) {
    return (u & 0x80000000u) ? __uint_as_float(u & 0x7fffffffu)
                             : __uint_as_float(~u);
}
#define ENC_NEG_INF 0x007FFFFFu   // enc(-inf)

__device__ __forceinline__ float lrelu(float x, float s) { return x > 0.f ? x : s * x; }

// ---------------------------------------------------------------------------
// 0) init: zero pooled output + scalar accumulators
// ---------------------------------------------------------------------------
__global__ void k_init_global(float* out, float* sc) {
    int i = blockIdx.x * blockDim.x + threadIdx.x;
    if (i < NGRAPH * 128) out[i] = 0.f;
    if (i < 8) sc[i] = 0.f;   // sc[0]=max bits (0 ok: attr>=0), [1]=sum_attr, [2]=sum_ea
}

// ---------------------------------------------------------------------------
// 1) edge_attr reductions: max + sum (attr >= 0 so raw-bit uint max is valid)
// ---------------------------------------------------------------------------
__global__ __launch_bounds__(256) void k_reduce_attr(const float* __restrict__ attr,
                                                     unsigned* maxbits, float* sumf) {
    __shared__ float smax[256], ssum[256];
    int t = threadIdx.x;
    int i = blockIdx.x * 256 + t;
    float v = (i < NEDGES) ? attr[i] : 0.f;
    smax[t] = v; ssum[t] = v; __syncthreads();
    for (int s = 128; s > 0; s >>= 1) {
        if (t < s) { smax[t] = fmaxf(smax[t], smax[t + s]); ssum[t] += ssum[t + s]; }
        __syncthreads();
    }
    if (t == 0) { atomicMax(maxbits, __float_as_uint(smax[0])); atomicAdd(sumf, ssum[0]); }
}

// ---------------------------------------------------------------------------
// 2) ea[e] = (a/mx) * exp(-|a/mx - mean/mx|), accumulate sum(ea)
// ---------------------------------------------------------------------------
__global__ __launch_bounds__(256) void k_compute_ea(const float* __restrict__ attr,
                                                    const float* __restrict__ sc,
                                                    float* __restrict__ ea, float* sum_ea) {
    __shared__ float ssum[256];
    int t = threadIdx.x;
    int i = blockIdx.x * 256 + t;
    float mx   = __uint_as_float(((const unsigned*)sc)[0]);
    float mean = sc[1] / (float)NEDGES;
    float v = 0.f;
    if (i < NEDGES) {
        float ew = attr[i] / mx;
        v = ew * __expf(-fabsf(ew - mean / mx));
        ea[i] = v;
    }
    ssum[t] = v; __syncthreads();
    for (int s = 128; s > 0; s >>= 1) { if (t < s) ssum[t] += ssum[t + s]; __syncthreads(); }
    if (t == 0) atomicAdd(sum_ea, ssum[0]);
}

// ---------------------------------------------------------------------------
// 3) fp32 -> bf16 elementwise convert
// ---------------------------------------------------------------------------
__global__ void k_f32_to_bf16(const float* __restrict__ in, __bf16* __restrict__ out, int n) {
    int i = blockIdx.x * blockDim.x + threadIdx.x;
    if (i < n) out[i] = (__bf16)in[i];
}

// ---------------------------------------------------------------------------
// 4) pack W[K,F] fp32 -> bf16 fragment layout: Wp[(kb*F + n)*16 + i] = W[kb*16+i][n]
//    so each lane's B fragment is 32 contiguous bytes.
// ---------------------------------------------------------------------------
__global__ void k_pack_w(const float* __restrict__ W, __bf16* __restrict__ Wp, int K, int F) {
    int idx = blockIdx.x * blockDim.x + threadIdx.x;
    if (idx >= K * F) return;
    int i  = idx & 15;
    int r  = idx >> 4;
    int n  = r % F;
    int kb = r / F;
    Wp[idx] = (__bf16)W[(kb * 16 + i) * F + n];
}

// ---------------------------------------------------------------------------
// 5) WMMA GEMM: H[N,F] = A[N,K](bf16, row-major) x Wp (packed bf16), f32 accum.
//    Block = 128 threads (4 waves). Block owns a 16-row strip; wave w owns
//    TILES 16-col tiles (F = 64*TILES). TILES/K are compile-time so the
//    accumulators stay in statically-named VGPRs (no v_movrel indexing).
// ---------------------------------------------------------------------------
template <int K, int TILES>
__global__ __launch_bounds__(128)
void k_gemm_wmma(const __bf16* __restrict__ A, const __bf16* __restrict__ Wp,
                 float* __restrict__ Hout) {
    constexpr int F = 64 * TILES;                  // 256 (TILES=4) or 128 (TILES=2)
    const int lane  = threadIdx.x & 31;
    const int wave  = threadIdx.x >> 5;            // 0..3
    const int m0    = blockIdx.x * 16;             // 16-row strip (N % 16 == 0)
    const int n0w   = wave * (16 * TILES);
    const int half  = lane >> 4;                   // 0/1
    const int lm    = lane & 15;
    const size_t m  = (size_t)(m0 + lm);

    v8f acc[TILES];
    #pragma unroll
    for (int t = 0; t < TILES; ++t) acc[t] = (v8f){0.f,0.f,0.f,0.f,0.f,0.f,0.f,0.f};

    const __bf16* arow = A + m * K + half * 16;
    const __bf16* bbase = Wp + ((size_t)half * F + n0w + lm) * 16;

    #pragma unroll
    for (int ks = 0; ks < K / 32; ++ks) {
        const int k0 = ks * 32;
        // A fragment: lane holds row m, K-run [k0+half*16, +16)
        Frag16 af;
        af.q[0] = *reinterpret_cast<const uint4*>(arow + k0);
        af.q[1] = *reinterpret_cast<const uint4*>(arow + k0 + 8);
        const __bf16* bp = bbase + (size_t)(2 * ks) * F * 16;  // kb = 2*ks + half folded in
        #pragma unroll
        for (int t = 0; t < TILES; ++t) {
            Frag16 bf;
            const __bf16* bpt = bp + (size_t)t * 16 * 16;
            bf.q[0] = reinterpret_cast<const uint4*>(bpt)[0];
            bf.q[1] = reinterpret_cast<const uint4*>(bpt)[1];
            acc[t] = __builtin_amdgcn_wmma_f32_16x16x32_bf16(
                false, af.v, false, bf.v, (short)0, acc[t], false, false);
        }
    }
    // D layout: vgpr r -> row m0 + r + 8*half, col n0 + lm
    #pragma unroll
    for (int t = 0; t < TILES; ++t) {
        const int n0 = n0w + t * 16;
        float* drow = Hout + (size_t)(m0 + half * 8) * F + n0 + lm;
        #pragma unroll
        for (int r = 0; r < 8; ++r) drow[(size_t)r * F] = acc[t][r];
    }
}

// ---------------------------------------------------------------------------
// 6) per-node attention coefficients: als/ald[n,h] = dot(h[n,h,:], a_src/dst[h,:])
//    one wave per node, shfl_xor tree reduce inside 32/H lane groups
// ---------------------------------------------------------------------------
__global__ __launch_bounds__(256)
void k_attn_coeff(const float* __restrict__ Hm, const float* __restrict__ a_src,
                  const float* __restrict__ a_dst, float* __restrict__ als,
                  float* __restrict__ ald, int F, int Hh) {
    int wid  = (blockIdx.x * blockDim.x + threadIdx.x) >> 5;
    int lane = threadIdx.x & 31;
    if (wid >= NNODES) return;
    int per = F >> 5;                              // 8 (F=256) or 4 (F=128)
    const float* hrow = Hm + (size_t)wid * F + lane * per;
    float s = 0.f, d = 0.f;
    for (int j = 0; j < per; ++j) {
        float hv = hrow[j];
        s += hv * a_src[lane * per + j];
        d += hv * a_dst[lane * per + j];
    }
    int lph = 32 / Hh;                             // lanes per head
    for (int off = 1; off < lph; off <<= 1) {
        s += __shfl_xor(s, off, 32);
        d += __shfl_xor(d, off, 32);
    }
    if ((lane & (lph - 1)) == 0) {
        int h = lane / lph;
        als[wid * Hh + h] = s;
        ald[wid * Hh + h] = d;
    }
}

// ---------------------------------------------------------------------------
// 7) ce[h] = dot(We[h,:], ae[h,:])  (al_e[e,h] = ea[e] * ce[h])
// ---------------------------------------------------------------------------
__global__ void k_compute_ce(const float* __restrict__ We, const float* __restrict__ ae,
                             float* ce, int Hh, int C) {
    int t = threadIdx.x;
    if (t < Hh) ce[t] = 0.f;
    __syncthreads();
    if (t < Hh * C) atomicAdd(&ce[t / C], We[t] * ae[t]);
}

// ---------------------------------------------------------------------------
// 8) per-layer attention init
// ---------------------------------------------------------------------------
__global__ void k_attn_init(unsigned* __restrict__ amax, float* __restrict__ denom,
                            float* __restrict__ outacc, int nh, int nout) {
    int i = blockIdx.x * blockDim.x + threadIdx.x;
    if (i < nh) { amax[i] = ENC_NEG_INF; denom[i] = 0.f; }
    if (i < nout) outacc[i] = 0.f;
}

// ---------------------------------------------------------------------------
// 9) edge pass 1: araw = leaky_relu(als[src]+ald[dst]+ea*ce, 0.2); segment max
// ---------------------------------------------------------------------------
__global__ __launch_bounds__(256)
void k_edge_pass1(const int* __restrict__ ei, int Hh, const float* __restrict__ ea,
                  const float* __restrict__ sc, const float* __restrict__ ce,
                  const float* __restrict__ als, const float* __restrict__ ald,
                  float* __restrict__ araw, unsigned* __restrict__ amax) {
    int t = blockIdx.x * blockDim.x + threadIdx.x;
    if (t >= NETOT * Hh) return;
    int e = t / Hh, h = t - e * Hh;
    int s, d; float eav;
    if (e < NEDGES) { s = ei[e]; d = ei[NEDGES + e]; eav = ea[e]; }
    else            { s = d = e - NEDGES; eav = sc[2] / (float)NEDGES; }
    float a = als[s * Hh + h] + ald[d * Hh + h] + eav * ce[h];
    a = lrelu(a, 0.2f);
    araw[t] = a;
    atomicMax(&amax[d * Hh + h], enc_f32(a));
}

// ---------------------------------------------------------------------------
// 10) edge pass 2: alpha = exp(araw - amax[dst]); segment sum denom
// ---------------------------------------------------------------------------
__global__ __launch_bounds__(256)
void k_edge_pass2(const int* __restrict__ ei, int Hh,
                  const unsigned* __restrict__ amax, float* __restrict__ araw,
                  float* __restrict__ denom) {
    int t = blockIdx.x * blockDim.x + threadIdx.x;
    if (t >= NETOT * Hh) return;
    int e = t / Hh, h = t - e * Hh;
    int d = (e < NEDGES) ? ei[NEDGES + e] : (e - NEDGES);
    float a = __expf(araw[t] - dec_f32(amax[d * Hh + h]));
    araw[t] = a;
    atomicAdd(&denom[d * Hh + h], a);
}

// ---------------------------------------------------------------------------
// 11) edge pass 3: out[dst] += (alpha/denom) * h[src]   (float4 gathers, f32 atomics)
// ---------------------------------------------------------------------------
__global__ __launch_bounds__(256)
void k_edge_pass3(const int* __restrict__ ei, int Hh, int C, int F,
                  const float* __restrict__ araw, const float* __restrict__ denom,
                  const float* __restrict__ Hm, float* __restrict__ outacc) {
    const int chunks = F >> 2;
    long long t = (long long)blockIdx.x * blockDim.x + threadIdx.x;
    if (t >= (long long)NETOT * chunks) return;
    int e  = (int)(t / chunks);
    int c4 = (int)(t - (long long)e * chunks);
    int f  = c4 * 4;
    int h  = f / C;
    int s, d;
    if (e < NEDGES) { s = ei[e]; d = ei[NEDGES + e]; }
    else            { s = d = e - NEDGES; }
    float w = araw[(size_t)e * Hh + h] / (denom[d * Hh + h] + 1e-16f);
    float4 hv = *reinterpret_cast<const float4*>(Hm + (size_t)s * F + f);
    float* o = outacc + (size_t)d * F + f;
    atomicAdd(o + 0, w * hv.x);
    atomicAdd(o + 1, w * hv.y);
    atomicAdd(o + 2, w * hv.z);
    atomicAdd(o + 3, w * hv.w);
}

// ---------------------------------------------------------------------------
// 12) epilogue (layers 1-3): x_next = bf16(leaky_relu(out + b, 0.01))
// ---------------------------------------------------------------------------
__global__ void k_epilogue_act(const float* __restrict__ outacc, const float* __restrict__ b,
                               __bf16* __restrict__ xnext, int F) {
    long long i = (long long)blockIdx.x * blockDim.x + threadIdx.x;
    if (i >= (long long)NNODES * F) return;
    float v = outacc[i] + b[(int)(i % F)];
    xnext[i] = (__bf16)lrelu(v, 0.01f);
}

// ---------------------------------------------------------------------------
// 13) epilogue layer 4 + global_add_pool: out[batch[n],f] += out4[n,f] + b4[f]
// ---------------------------------------------------------------------------
__global__ void k_epilogue_pool(const float* __restrict__ outacc, const float* __restrict__ b4,
                                const int* __restrict__ batch, float* __restrict__ out, int F) {
    long long i = (long long)blockIdx.x * blockDim.x + threadIdx.x;
    if (i >= (long long)NNODES * F) return;
    int n = (int)(i / F), f = (int)(i - (long long)n * F);
    float v = outacc[i] + b4[f];
    atomicAdd(&out[batch[n] * F + f], v);
}

// ---------------------------------------------------------------------------
// host launch
// ---------------------------------------------------------------------------
static inline int cdiv(long long a, long long b) { return (int)((a + b - 1) / b); }

extern "C" void kernel_launch(void* const* d_in, const int* in_sizes, int n_in,
                              void* d_out, int out_size, void* d_ws, size_t ws_size,
                              hipStream_t stream) {
    const float* x         = (const float*)d_in[0];
    const int*   ei        = (const int*)  d_in[1];
    const float* edge_attr = (const float*)d_in[2];
    const int*   batch     = (const int*)  d_in[3];

    struct LayerP { const float *W,*as_,*ad_,*We,*ae,*b; int H,C,Fin,Fout; };
    LayerP L[4];
    for (int l = 0; l < 4; ++l) {
        int base = 4 + l * 6;
        L[l] = { (const float*)d_in[base+0], (const float*)d_in[base+1],
                 (const float*)d_in[base+2], (const float*)d_in[base+3],
                 (const float*)d_in[base+4], (const float*)d_in[base+5],
                 4, 64, 256, 256 };
    }
    L[0].Fin = 128;
    L[3].H = 1; L[3].C = 128; L[3].Fout = 128;

    // workspace carve-out (~148 MB)
    char* p = (char*)d_ws;
    auto carve = [&](size_t bytes) { char* r = p; p += (bytes + 255) & ~(size_t)255; return r; };
    float*    sc     = (float*)   carve(256);                          // scalars
    float*    ce     = sc + 4;                                         // ce[0..3]
    float*    ea     = (float*)   carve((size_t)NEDGES * 4);
    __bf16*   xb     = (__bf16*)  carve((size_t)NNODES * 256 * 2);
    __bf16*   Wp     = (__bf16*)  carve((size_t)256 * 256 * 2);
    float*    hbuf   = (float*)   carve((size_t)NNODES * 256 * 4);
    float*    als    = (float*)   carve((size_t)NNODES * 4 * 4);
    float*    ald    = (float*)   carve((size_t)NNODES * 4 * 4);
    unsigned* amax   = (unsigned*)carve((size_t)NNODES * 4 * 4);
    float*    denom  = (float*)   carve((size_t)NNODES * 4 * 4);
    float*    araw   = (float*)   carve((size_t)NETOT  * 4 * 4);
    float*    outacc = (float*)   carve((size_t)NNODES * 256 * 4);
    (void)ws_size;

    float* out = (float*)d_out;

    // ---- global init + edge preprocessing ----
    k_init_global<<<cdiv(NGRAPH * 128, 256), 256, 0, stream>>>(out, sc);
    k_reduce_attr<<<cdiv(NEDGES, 256), 256, 0, stream>>>(edge_attr, (unsigned*)&sc[0], &sc[1]);
    k_compute_ea<<<cdiv(NEDGES, 256), 256, 0, stream>>>(edge_attr, sc, ea, &sc[2]);

    // ---- x -> bf16 for layer 1 ----
    k_f32_to_bf16<<<cdiv((long long)NNODES * 128, 256), 256, 0, stream>>>(x, xb, NNODES * 128);

    for (int l = 0; l < 4; ++l) {
        const LayerP& q = L[l];
        const int F = q.Fout, K = q.Fin, H = q.H, C = q.C;

        k_pack_w<<<cdiv(K * F, 256), 256, 0, stream>>>(q.W, Wp, K, F);

        if      (l == 0) k_gemm_wmma<128, 4><<<NNODES / 16, 128, 0, stream>>>(xb, Wp, hbuf);
        else if (l == 3) k_gemm_wmma<256, 2><<<NNODES / 16, 128, 0, stream>>>(xb, Wp, hbuf);
        else             k_gemm_wmma<256, 4><<<NNODES / 16, 128, 0, stream>>>(xb, Wp, hbuf);

        k_attn_coeff<<<cdiv((long long)NNODES * 32, 256), 256, 0, stream>>>(
            hbuf, q.as_, q.ad_, als, ald, F, H);
        k_compute_ce<<<1, 256, 0, stream>>>(q.We, q.ae, ce, H, C);
        k_attn_init<<<cdiv((long long)NNODES * F, 256), 256, 0, stream>>>(
            amax, denom, outacc, NNODES * H, NNODES * F);
        k_edge_pass1<<<cdiv((long long)NETOT * H, 256), 256, 0, stream>>>(
            ei, H, ea, sc, ce, als, ald, araw, amax);
        k_edge_pass2<<<cdiv((long long)NETOT * H, 256), 256, 0, stream>>>(
            ei, H, amax, araw, denom);
        k_edge_pass3<<<cdiv((long long)NETOT * (F / 4), 256), 256, 0, stream>>>(
            ei, H, C, F, araw, denom, hbuf, outacc);

        if (l < 3) {
            k_epilogue_act<<<cdiv((long long)NNODES * F, 256), 256, 0, stream>>>(
                outacc, q.b, xb, F);
        } else {
            k_epilogue_pool<<<cdiv((long long)NNODES * F, 256), 256, 0, stream>>>(
                outacc, q.b, batch, out, F);
        }
    }
}
